// NMBellmanFordLayer_3745211482943
// MI455X (gfx1250) — matrix operand order repaired and checked by reading, not compiled, not run on hardware
//
#include <hip/hip_runtime.h>

typedef __attribute__((ext_vector_type(16))) _Float16 v16h;
typedef __attribute__((ext_vector_type(8)))  _Float16 v8h;
typedef __attribute__((ext_vector_type(8)))  float    v8f;

#define DD     128          // feature dim D
#define TILE_E 256          // edges per workgroup tile (32 per wave, 8 waves)
#define TILE_M 128          // node rows per workgroup tile
#define PH     (DD + 8)     // padded f16 pitch, 128-wide (136 halves = 272B)
#define PX     (2*DD + 8)   // padded f16 pitch, 256-wide (264 halves)
#define PO     (DD + 1)     // padded f32 pitch for LN scratch
#define LN_EPS 1e-5f

union V16 { v16h v; v8h h[2]; };

__device__ __forceinline__ v8f wmma_f16(v16h a, v16h b, v8f c) {
  // D = A(16x32 f16) * B(32x16 f16) + C(16x16 f32)
  return __builtin_amdgcn_wmma_f32_16x16x32_f16(
      /*neg_a=*/false, a, /*neg_b=*/false, b,
      /*c_mod=*/(short)0, c, /*reuse_a=*/false, /*reuse_b=*/false);
}

__device__ __forceinline__ float fast_sigmoid(float x) {
  // v_exp_f32 + v_add + v_rcp_f32 (avoids IEEE divide expansion)
  return __builtin_amdgcn_rcpf(1.0f + __expf(-x));
}

// ---------------------------------------------------------------------------
// Phase 0: zero h_agg accumulator in workspace
// ---------------------------------------------------------------------------
__global__ void zero_kernel(float4* __restrict__ p, long n4) {
  long i = (long)blockIdx.x * blockDim.x + threadIdx.x;
  long stride = (long)gridDim.x * blockDim.x;
  for (; i < n4; i += stride) p[i] = make_float4(0.f, 0.f, 0.f, 0.f);
}

// ---------------------------------------------------------------------------
// Phase 1: per-edge  gate = sigmoid(h_src @ gate_W),
//                    msg  = (h_src @ msg_W) * rel_emb[rel] * gate,
//                    h_agg[:, tgt] += msg         (no-return f32 atomics in L2)
// GEMM per batch: (E x 128) x (128 x 128), f16 WMMA, f32 accumulate.
// Each wave register-blocks M=32 (two A tile-sets) so every B fragment feeds
// two WMMAs -> ~1.1 ds_load_b128 per WMMA (LDS bandwidth balanced).
// ---------------------------------------------------------------------------
__global__ __launch_bounds__(256) void edge_kernel(
    const float* __restrict__ h, const int* __restrict__ eSrc,
    const int* __restrict__ eTgt, const int* __restrict__ eRel,
    const float* __restrict__ msgW, const float* __restrict__ relEmb,
    const float* __restrict__ gateW, float* __restrict__ hAgg,
    int Nn, int Ee, int Bb, int nTilesE)
{
  __shared__ __align__(16) _Float16 sGT[DD * PH];     // gate_W^T  [n][k]
  __shared__ __align__(16) _Float16 sMT[DD * PH];     // msg_W^T   [n][k]
  __shared__ __align__(16) _Float16 sH [TILE_E * PH]; // gathered h_src rows (f16)
  __shared__ int sSrc[TILE_E], sTgt[TILE_E], sRel[TILE_E];

  const int tid = threadIdx.x;

  // stage both weight matrices transposed, f32 -> f16 (once per block)
  for (int i = tid; i < DD * DD; i += 256) {
    int k = i >> 7, n = i & (DD - 1);
    sGT[n * PH + k] = (_Float16)gateW[i];
    sMT[n * PH + k] = (_Float16)msgW[i];
  }

  const int wave = tid >> 5, lane = tid & 31;
  const int hh = lane >> 4, nl = lane & 15;   // wave half, lane-in-half
  const int rowBase = wave * 32;              // 32 edges per wave

  const int totalTiles = Bb * nTilesE;
  for (int t = blockIdx.x; t < totalTiles; t += gridDim.x) {
    const int b  = t / nTilesE;
    const int e0 = (t - b * nTilesE) * TILE_E;

    __syncthreads();   // weights staged / previous tile fully consumed
    {
      int e = e0 + tid;
      sSrc[tid] = (e < Ee) ? eSrc[e] : -1;
      sTgt[tid] = (e < Ee) ? eTgt[e] : -1;
      sRel[tid] = (e < Ee) ? eRel[e] : 0;
    }
    __syncthreads();

    // gather 256 source rows, convert to f16 in LDS (vector f32x4 loads)
    for (int i = tid; i < TILE_E * (DD / 4); i += 256) {
      int row = i >> 5, kq = i & 31;
      int s = sSrc[row];
      float4 v = make_float4(0.f, 0.f, 0.f, 0.f);
      if (s >= 0) v = ((const float4*)(h + ((size_t)b * Nn + s) * DD))[kq];
      _Float16* dst = &sH[row * PH + kq * 4];
      dst[0] = (_Float16)v.x; dst[1] = (_Float16)v.y;
      dst[2] = (_Float16)v.z; dst[3] = (_Float16)v.w;
    }
    __syncthreads();

    // A fragments: two 16-row tile-sets (rows rowBase+0..15 and +16..31)
    V16 A0[4], A1[4];
    const int am0 = rowBase + nl;
    const int am1 = rowBase + 16 + nl;
    #pragma unroll
    for (int kb = 0; kb < 4; ++kb) {
      const v8h* p0 = (const v8h*)(sH + am0 * PH + kb * 32 + hh * 16);
      A0[kb].h[0] = p0[0];   // K = 32kb + {0..7}  (+16 upper half)
      A0[kb].h[1] = p0[1];   // K = 32kb + {8..15} (+16 upper half)
      const v8h* p1 = (const v8h*)(sH + am1 * PH + kb * 32 + hh * 16);
      A1[kb].h[0] = p1[0];
      A1[kb].h[1] = p1[1];
    }

    #pragma unroll 2
    for (int nt = 0; nt < 8; ++nt) {
      v8f cg0 = {}, cg1 = {}, cm0 = {}, cm1 = {};
      const int bn = nt * 16 + nl;   // output column = W^T row
      #pragma unroll
      for (int kb = 0; kb < 4; ++kb) {
        V16 Bg, Bm;
        const v8h* pg = (const v8h*)(sGT + bn * PH + kb * 32 + hh * 16);
        Bg.h[0] = pg[0]; Bg.h[1] = pg[1];
        cg0 = wmma_f16(A0[kb].v, Bg.v, cg0);
        cg1 = wmma_f16(A1[kb].v, Bg.v, cg1);
        const v8h* pm = (const v8h*)(sMT + bn * PH + kb * 32 + hh * 16);
        Bm.h[0] = pm[0]; Bm.h[1] = pm[1];
        cm0 = wmma_f16(A0[kb].v, Bm.v, cm0);
        cm1 = wmma_f16(A1[kb].v, Bm.v, cm1);
      }
      // epilogue: C VGPR r <-> row M = r + 8*half, col N = nt*16 + (lane&15)
      const int col = nt * 16 + nl;
      #pragma unroll
      for (int r = 0; r < 8; ++r) {
        const int li0 = rowBase + r + hh * 8;
        const int tg0 = sTgt[li0];
        if (tg0 >= 0) {
          float gate = fast_sigmoid(cg0[r]);
          float rv = relEmb[(size_t)sRel[li0] * DD + col];
          unsafeAtomicAdd(&hAgg[((size_t)b * Nn + tg0) * DD + col],
                          cm0[r] * rv * gate);
        }
        const int li1 = li0 + 16;
        const int tg1 = sTgt[li1];
        if (tg1 >= 0) {
          float gate = fast_sigmoid(cg1[r]);
          float rv = relEmb[(size_t)sRel[li1] * DD + col];
          unsafeAtomicAdd(&hAgg[((size_t)b * Nn + tg1) * DD + col],
                          cm1[r] * rv * gate);
        }
      }
    }
  }
}

// ---------------------------------------------------------------------------
// Phase 2: upd = concat(h, h_agg) @ update_W + b ; x = h + relu(upd) ; LN(x)
// GEMM: (B*N x 256) x (256 x 128), f16 WMMA, f32 accumulate.
// ---------------------------------------------------------------------------
__global__ __launch_bounds__(256) void node_kernel(
    const float* __restrict__ h, const float* __restrict__ hAgg,
    const float* __restrict__ updW, const float* __restrict__ updB,
    const float* __restrict__ lnG, const float* __restrict__ lnB,
    float* __restrict__ out, int Mtot, int nTilesM)
{
  __shared__ __align__(16) _Float16 sWT[DD * PX];      // update_W^T [n][k=0..255]
  __shared__ __align__(16) _Float16 sX [TILE_M * PX];  // concat rows (f16)
  __shared__ __align__(16) float    sO [TILE_M * PO];  // x rows for LayerNorm

  const int tid = threadIdx.x;
  for (int i = tid; i < 2 * DD * DD; i += 256) {
    int k = i >> 7, n = i & (DD - 1);
    sWT[n * PX + k] = (_Float16)updW[i];
  }

  const int wave = tid >> 5, lane = tid & 31;
  const int hh = lane >> 4, nl = lane & 15;
  const int rowBase = wave * 16;

  for (int t = blockIdx.x; t < nTilesM; t += gridDim.x) {
    const int g0 = t * TILE_M;
    __syncthreads();
    // stage concat(h, h_agg) rows as f16
    for (int i = tid; i < TILE_M * (2 * DD / 4); i += 256) {
      int row = i >> 6, cq = i & 63;
      int g = g0 + row;
      float4 v = make_float4(0.f, 0.f, 0.f, 0.f);
      if (g < Mtot) {
        if (cq < 32) v = ((const float4*)(h    + (size_t)g * DD))[cq];
        else         v = ((const float4*)(hAgg + (size_t)g * DD))[cq - 32];
      }
      _Float16* dst = &sX[row * PX + cq * 4];
      dst[0] = (_Float16)v.x; dst[1] = (_Float16)v.y;
      dst[2] = (_Float16)v.z; dst[3] = (_Float16)v.w;
    }
    __syncthreads();

    V16 A[8];
    const int am = rowBase + nl;
    #pragma unroll
    for (int kb = 0; kb < 8; ++kb) {
      const v8h* p = (const v8h*)(sX + am * PX + kb * 32 + hh * 16);
      A[kb].h[0] = p[0];
      A[kb].h[1] = p[1];
    }
    #pragma unroll 2
    for (int nt = 0; nt < 8; ++nt) {
      v8f c = {};
      const int bn = nt * 16 + nl;
      #pragma unroll
      for (int kb = 0; kb < 8; ++kb) {
        V16 Bw;
        const v8h* pw = (const v8h*)(sWT + bn * PX + kb * 32 + hh * 16);
        Bw.h[0] = pw[0]; Bw.h[1] = pw[1];
        c = wmma_f16(A[kb].v, Bw.v, c);
      }
      const int col = nt * 16 + nl;
      #pragma unroll
      for (int r = 0; r < 8; ++r) {
        const int li = rowBase + r + hh * 8;
        const int g = g0 + li;
        if (g < Mtot) {
          float upd = c[r] + updB[col];
          float x = h[(size_t)g * DD + col] + fmaxf(upd, 0.f);
          sO[li * PO + col] = x;
        }
      }
    }
    __syncthreads();
    // LayerNorm: one thread per row (padded pitch avoids bank serialization)
    if (tid < TILE_M) {
      int g = g0 + tid;
      if (g < Mtot) {
        float mu = 0.f;
        #pragma unroll 4
        for (int k = 0; k < DD; ++k) mu += sO[tid * PO + k];
        mu *= (1.0f / DD);
        float var = 0.f;
        #pragma unroll 4
        for (int k = 0; k < DD; ++k) {
          float d = sO[tid * PO + k] - mu; var += d * d;
        }
        var *= (1.0f / DD);
        float inv = rsqrtf(var + LN_EPS);
        for (int k = 0; k < DD; ++k) {
          out[(size_t)g * DD + k] =
              (sO[tid * PO + k] - mu) * inv * lnG[k] + lnB[k];
        }
      }
    }
  }
}

// ---------------------------------------------------------------------------
extern "C" void kernel_launch(void* const* d_in, const int* in_sizes, int n_in,
                              void* d_out, int out_size, void* d_ws, size_t ws_size,
                              hipStream_t stream) {
  const float* h      = (const float*)d_in[0];
  const int*   eSrc   = (const int*)d_in[1];
  const int*   eTgt   = (const int*)d_in[2];
  const int*   eRel   = (const int*)d_in[3];
  /* d_in[4] = nE (device scalar) — N derived from shapes instead */
  const float* msgW   = (const float*)d_in[5];
  const float* relEmb = (const float*)d_in[6];
  const float* gateW  = (const float*)d_in[7];
  const float* updW   = (const float*)d_in[8];
  const float* updB   = (const float*)d_in[9];
  const float* lnG    = (const float*)d_in[10];
  const float* lnB    = (const float*)d_in[11];
  float* out  = (float*)d_out;
  float* hAgg = (float*)d_ws;                 // B*N*D floats of scratch

  const int B = 2;                            // fixed by reference
  const int N = in_sizes[0] / (B * DD);
  const int E = in_sizes[1];

  // Phase 0: zero the aggregation buffer (must happen every call)
  long n4 = (long)in_sizes[0] / 4;
  int zb = (int)((n4 + 255) / 256); if (zb > 16384) zb = 16384;
  zero_kernel<<<zb, 256, 0, stream>>>((float4*)hAgg, n4);

  // Phase 1: edge GEMMs + gated scatter-add
  int nTilesE = (E + TILE_E - 1) / TILE_E;
  int eb = B * nTilesE; if (eb > 2048) eb = 2048;
  edge_kernel<<<eb, 256, 0, stream>>>(h, eSrc, eTgt, eRel, msgW, relEmb,
                                      gateW, hAgg, N, E, B, nTilesE);

  // Phase 2: node update GEMM + residual ReLU + LayerNorm
  int Mtot = B * N;
  int nTilesM = (Mtot + TILE_M - 1) / TILE_M;
  int nb = nTilesM; if (nb > 2048) nb = 2048;
  node_kernel<<<nb, 256, 0, stream>>>(h, hAgg, updW, updB, lnG, lnB,
                                      out, Mtot, nTilesM);
}
